// ImageMoE_37623913513503
// MI455X (gfx1250) — matrix operand order repaired and checked by reading, not compiled, not run on hardware
//
#include <hip/hip_runtime.h>
#include <math.h>

typedef __bf16 v16bf __attribute__((ext_vector_type(16)));
typedef float  v8f   __attribute__((ext_vector_type(8)));

#define T_TOK   8192      // B*S
#define D_MODEL 512
#define HID_DIM 2048
#define SEQ     256
#define NBATCH  32
#define NHEAD   8
#define HD      64
#define BHEADS  256       // B*H
#define NEXP    8
#define PD      588
#define PDP     640       // padded to multiple of 64 (K-step)

// gfx1250 async global->LDS staging (ASYNCcnt path), if the toolchain exposes it
#if defined(__gfx1250__)
#if __has_builtin(__builtin_amdgcn_global_load_async_to_lds_b128) && \
    __has_builtin(__builtin_amdgcn_s_wait_asynccnt)
#define USE_ASYNC_LDS 1
#endif
#endif

__device__ __forceinline__ __bf16 f2bf(float f) {
  unsigned u = __builtin_bit_cast(unsigned, f);
  unsigned r = (u + 0x7FFFu + ((u >> 16) & 1u)) >> 16;
  unsigned short s = (unsigned short)r;
  return __builtin_bit_cast(__bf16, s);
}
__device__ __forceinline__ float bf2f(__bf16 b) { return (float)b; }

// ---------------------------------------------------------------- zero fill
__global__ void k_zero(float* p, long long n) {
  long long i = (long long)blockIdx.x * blockDim.x + threadIdx.x;
  if (i < n) p[i] = 0.f;
}

// ------------------------------------------------- patch extraction -> bf16
// out[t][col], col=(p1*14+p2)*3+c, zero padded to PDP
__global__ void k_patchify(const float* __restrict__ x, __bf16* __restrict__ out) {
  long long i = (long long)blockIdx.x * blockDim.x + threadIdx.x;
  if (i >= (long long)T_TOK * PDP) return;
  int col = (int)(i % PDP);
  long long t = i / PDP;
  float val = 0.f;
  if (col < PD) {
    int c = col % 3, pp = col / 3;
    int p1 = pp / 14, p2 = pp % 14;
    int b = (int)(t / SEQ), hw = (int)(t % SEQ);
    int hh = hw / 16, ww = hw % 16;
    val = x[(((long long)b * 3 + c) * 224 + hh * 14 + p1) * 224 + ww * 14 + p2];
  }
  out[i] = f2bf(val);
}

// ------------------- weight convert: fp32 [batch][K][N] -> bf16 [batch][N][Kpad]
__global__ void k_wtrans(const float* __restrict__ w, __bf16* __restrict__ out,
                         int K, int N, int Kpad, long long total) {
  long long i = (long long)blockIdx.x * blockDim.x + threadIdx.x;
  if (i >= total) return;
  int k = (int)(i % Kpad);
  long long rest = i / Kpad;
  int n = (int)(rest % N);
  long long b = rest / N;
  float v = (k < K) ? w[(b * K + k) * (long long)N + n] : 0.f;
  out[i] = f2bf(v);
}

// ---------------------------------------------------------------- WMMA GEMM
// C[b] = accum*C + rowScale[row]*( alpha*(A[b] @ Bt[b]^T) + bias[b][col] )
// A: [batch][M][lda] bf16 row-major, Bt: [batch][N][ldb] bf16 (N-major), C fp32.
// Block tile 256(M) x 64(N), K-step 64. 8 waves, each wave owns 32x64 (2x4 WMMA).
// Requires M%256==0, N%64==0, K%64==0.
#define LDP 72   // padded LDS row stride (elems): 144B, conflict-free, 16B aligned
__global__ __launch_bounds__(256) void k_gemm_bf16(
    const __bf16* __restrict__ A, long long sA, int lda,
    const __bf16* __restrict__ Bt, long long sB, int ldb,
    float* __restrict__ C, long long sC, int ldc,
    const float* __restrict__ bias, long long sBias,
    const float* __restrict__ rowScale, int rsStride,
    float alpha, int accum, int K) {
  __shared__ __align__(16) __bf16 As[256][LDP];  // 36.9 KB
  __shared__ __align__(16) __bf16 Bs[64][LDP];   //  9.2 KB

  const int batch = blockIdx.z;
  const __bf16* Ab = A + (long long)batch * sA;
  const __bf16* Bb = Bt + (long long)batch * sB;
  float* Cb = C + (long long)batch * sC;
  const float* biasb = bias ? (bias + (long long)batch * sBias) : nullptr;

  const int m0 = blockIdx.x * 256;
  const int n0 = blockIdx.y * 64;
  const int tid = threadIdx.x;
  const int lane = tid & 31;
  const int wave = tid >> 5;
  const int wm = wave << 5;         // wave's 32-row slice
  const int half = lane >> 4;
  const int lrow = lane & 15;

  const int br = tid >> 2, bc = (tid & 3) * 16;  // B coop load: 4 thr/row

  v8f acc[2][4] = {};

  const __bf16* agRow = Ab + (long long)(m0 + tid) * lda;  // A coop: 1 thr/row
  const __bf16* bgRow = Bb + (long long)(n0 + br) * ldb + bc;

  for (int k0 = 0; k0 < K; k0 += 64) {
    const __bf16* ag = agRow + k0;
    const __bf16* bg = bgRow + k0;
#ifdef USE_ASYNC_LDS
    {
      // GLOBAL_LOAD_ASYNC_TO_LDS_B128: VGPR-free DMA into LDS, ASYNCcnt-tracked.
      // INST_OFFSET is added to BOTH the global and LDS addresses (ISA 10.x),
      // so one base pair + literal offsets stages a 128B row.
      typedef int v4si __attribute__((__vector_size__(16)));
      typedef __attribute__((address_space(1))) v4si GV4;
      typedef __attribute__((address_space(3))) v4si LV4;
      GV4* agp = (GV4*)(size_t)ag;
      LV4* alp = (LV4*)(unsigned)(size_t)&As[tid][0];
#define A_ASYNC(off) __builtin_amdgcn_global_load_async_to_lds_b128(agp, alp, off, 0)
      A_ASYNC(0);  A_ASYNC(16); A_ASYNC(32); A_ASYNC(48);
      A_ASYNC(64); A_ASYNC(80); A_ASYNC(96); A_ASYNC(112);
#undef A_ASYNC
      GV4* bgp = (GV4*)(size_t)bg;
      LV4* blp = (LV4*)(unsigned)(size_t)&Bs[br][bc];
      __builtin_amdgcn_global_load_async_to_lds_b128(bgp, blp, 0, 0);
      __builtin_amdgcn_global_load_async_to_lds_b128(bgp, blp, 16, 0);
      if (k0 + 64 < K) {
        __builtin_prefetch(ag + 64, 0, 1);
        __builtin_prefetch(bg + 64, 0, 1);
      }
      __builtin_amdgcn_s_wait_asynccnt(0);
    }
#else
    {
      // Pipelined fallback: distinct temporaries so all global loads issue
      // before any ds_store drains (no per-chunk loadcnt-0 stalls).
      uint4 ta[8], tb[2];
#pragma unroll
      for (int c = 0; c < 8; c++) ta[c] = *(const uint4*)(ag + c * 8);
      tb[0] = *(const uint4*)(bg);
      tb[1] = *(const uint4*)(bg + 8);
      if (k0 + 64 < K) {
        __builtin_prefetch(ag + 64, 0, 1);
        __builtin_prefetch(bg + 64, 0, 1);
      }
#pragma unroll
      for (int c = 0; c < 8; c++) *(uint4*)&As[tid][c * 8] = ta[c];
      *(uint4*)&Bs[br][bc] = tb[0];
      *(uint4*)&Bs[br][bc + 8] = tb[1];
    }
#endif
    __syncthreads();

#pragma unroll
    for (int kk = 0; kk < 2; kk++) {  // two 32-wide K sub-steps
      const int ko = kk * 32;
      v16bf afr[2], bfr[4];
#pragma unroll
      for (int i = 0; i < 2; i++) {
        // A 16x32 bf16 layout: lane<16 -> K{0..7,16..23}; lane>=16 -> K{8..15,24..31}
        union { uint4 u[2]; v16bf v; } ua;
        const __bf16* ap = &As[wm + i * 16 + lrow][ko];
        ua.u[0] = *(const uint4*)(ap + half * 8);
        ua.u[1] = *(const uint4*)(ap + 16 + half * 8);
        afr[i] = ua.v;
      }
#pragma unroll
      for (int j = 0; j < 4; j++) {
        // B 32x16 bf16 layout: lane<16 -> K{0..7,16..23}; lane>=16 -> K{8..15,24..31} (col = lane%16)
        union { uint4 u[2]; v16bf v; } ub;
        const __bf16* bp = &Bs[j * 16 + lrow][ko];
        ub.u[0] = *(const uint4*)(bp + half * 8);
        ub.u[1] = *(const uint4*)(bp + 16 + half * 8);
        bfr[j] = ub.v;
      }
#pragma unroll
      for (int i = 0; i < 2; i++)
#pragma unroll
        for (int j = 0; j < 4; j++)
          acc[i][j] = __builtin_amdgcn_wmma_f32_16x16x32_bf16(
              false, afr[i], false, bfr[j], (short)0, acc[i][j], false, false);
    }
    __syncthreads();
  }

#pragma unroll
  for (int i = 0; i < 2; i++) {
#pragma unroll
    for (int j = 0; j < 4; j++) {
      const int col = n0 + j * 16 + lrow;
      const int rb = m0 + wm + i * 16 + half * 8;
      const float bsv = biasb ? biasb[col] : 0.f;
#pragma unroll
      for (int r = 0; r < 8; r++) {
        float v = alpha * acc[i][j][r] + bsv;
        const int row = rb + r;
        if (rowScale) v *= rowScale[(long long)row * rsStride];
        long long idx = (long long)row * ldc + col;
        if (accum) Cb[idx] += v; else Cb[idx] = v;
      }
    }
  }
}

// ------------------------------------------------------------ add pos embed
__global__ void k_addpos(float* __restrict__ e, const float* __restrict__ pos) {
  long long i = (long long)blockIdx.x * blockDim.x + threadIdx.x;
  if (i >= (long long)T_TOK * D_MODEL) return;
  int d = (int)(i & 511);
  int s = (int)((i >> 9) % SEQ);
  e[i] += pos[s * D_MODEL + d];
}

// ------------------------------------------------------ layernorm -> bf16
__global__ __launch_bounds__(256) void k_ln(const float* __restrict__ x,
                                            const float* __restrict__ g,
                                            const float* __restrict__ b,
                                            __bf16* __restrict__ out) {
  const int lane = threadIdx.x & 31;
  const long long t = (long long)blockIdx.x * 8 + (threadIdx.x >> 5);
  const float* xr = x + t * D_MODEL;
  float s = 0.f, s2 = 0.f, vals[16];
#pragma unroll
  for (int i = 0; i < 16; i++) {
    float v = xr[i * 32 + lane];
    vals[i] = v; s += v; s2 += v * v;
  }
#pragma unroll
  for (int m = 16; m >= 1; m >>= 1) {
    s += __shfl_xor(s, m, 32);
    s2 += __shfl_xor(s2, m, 32);
  }
  const float mean = s * (1.f / 512.f);
  const float var = s2 * (1.f / 512.f) - mean * mean;
  const float rs = rsqrtf(var + 1e-5f);
  __bf16* orow = out + t * D_MODEL;
#pragma unroll
  for (int i = 0; i < 16; i++) {
    int d = i * 32 + lane;
    orow[d] = f2bf((vals[i] - mean) * rs * g[d] + b[d]);
  }
}

// -------------- split heads: Q,K -> [bh][s][hd]; V -> transposed [bh][hd][s]
__global__ void k_split(const float* __restrict__ Q, const float* __restrict__ Kk,
                        const float* __restrict__ V, __bf16* __restrict__ qa,
                        __bf16* __restrict__ ka, __bf16* __restrict__ vt) {
  long long i = (long long)blockIdx.x * blockDim.x + threadIdx.x;
  if (i >= (long long)T_TOK * D_MODEL) return;
  int c = (int)(i & 511);
  long long t = i >> 9;
  int h = c >> 6, d = c & 63;
  int b = (int)(t / SEQ), s = (int)(t % SEQ);
  long long bh = (long long)b * NHEAD + h;
  qa[bh * SEQ * HD + (long long)s * HD + d] = f2bf(Q[i]);
  ka[bh * SEQ * HD + (long long)s * HD + d] = f2bf(Kk[i]);
  vt[bh * HD * SEQ + (long long)d * SEQ + s] = f2bf(V[i]);
}

// ---------------- softmax over rows of 256 (in place fp32) + bf16 shadow
__global__ __launch_bounds__(256) void k_softmax(float* __restrict__ sc,
                                                 __bf16* __restrict__ scb) {
  const int lane = threadIdx.x & 31;
  const long long row = (long long)blockIdx.x * 8 + (threadIdx.x >> 5);
  float* r = sc + row * SEQ;
  float v[8], mx = -1e30f;
#pragma unroll
  for (int i = 0; i < 8; i++) { v[i] = r[i * 32 + lane]; mx = fmaxf(mx, v[i]); }
#pragma unroll
  for (int m = 16; m >= 1; m >>= 1) mx = fmaxf(mx, __shfl_xor(mx, m, 32));
  float s = 0.f;
#pragma unroll
  for (int i = 0; i < 8; i++) { v[i] = expf(v[i] - mx); s += v[i]; }
#pragma unroll
  for (int m = 16; m >= 1; m >>= 1) s += __shfl_xor(s, m, 32);
  const float inv = 1.f / s;
#pragma unroll
  for (int i = 0; i < 8; i++) {
    float p = v[i] * inv;
    r[i * 32 + lane] = p;
    scb[row * SEQ + i * 32 + lane] = f2bf(p);
  }
}

// ----------- attn_w = softmax(mean over heads of attn) -> d_out section
__global__ __launch_bounds__(256) void k_attnw(const float* __restrict__ attn,
                                               float* __restrict__ outw) {
  const int lane = threadIdx.x & 31;
  const long long row = (long long)blockIdx.x * 8 + (threadIdx.x >> 5);  // b*SEQ+q
  const int b = (int)(row / SEQ), q = (int)(row % SEQ);
  float v[8], mx = -1e30f;
#pragma unroll
  for (int i = 0; i < 8; i++) {
    int col = i * 32 + lane;
    float s = 0.f;
    for (int h = 0; h < NHEAD; h++)
      s += attn[(((long long)b * NHEAD + h) * SEQ + q) * SEQ + col];
    v[i] = s * 0.125f;
    mx = fmaxf(mx, v[i]);
  }
#pragma unroll
  for (int m = 16; m >= 1; m >>= 1) mx = fmaxf(mx, __shfl_xor(mx, m, 32));
  float s = 0.f;
#pragma unroll
  for (int i = 0; i < 8; i++) { v[i] = expf(v[i] - mx); s += v[i]; }
#pragma unroll
  for (int m = 16; m >= 1; m >>= 1) s += __shfl_xor(s, m, 32);
  const float inv = 1.f / s;
#pragma unroll
  for (int i = 0; i < 8; i++) outw[row * SEQ + i * 32 + lane] = v[i] * inv;
}

// ---------------- merge heads: [bh][s][hd] fp32 -> [t][D] bf16
__global__ void k_merge(const float* __restrict__ ao, __bf16* __restrict__ merged) {
  long long i = (long long)blockIdx.x * blockDim.x + threadIdx.x;
  if (i >= (long long)T_TOK * D_MODEL) return;
  int c = (int)(i & 511);
  long long t = i >> 9;
  int h = c >> 6, d = c & 63;
  int b = (int)(t / SEQ), s = (int)(t % SEQ);
  merged[i] = f2bf(ao[(((long long)b * NHEAD + h) * SEQ + s) * HD + d]);
}

// --------------- router: probs, top-2 gates, aux-loss accumulators
__global__ __launch_bounds__(256) void k_router(const __bf16* __restrict__ x,
                                                const float* __restrict__ rw,
                                                const float* __restrict__ rb,
                                                float* __restrict__ gates,
                                                float* __restrict__ frac,
                                                float* __restrict__ imp) {
  long long t = (long long)blockIdx.x * blockDim.x + threadIdx.x;
  if (t >= T_TOK) return;
  float lg[NEXP];
#pragma unroll
  for (int e = 0; e < NEXP; e++) lg[e] = rb[e];
  const __bf16* xr = x + t * D_MODEL;
  for (int d = 0; d < D_MODEL; d++) {
    float xv = bf2f(xr[d]);
    const float* rwr = rw + d * NEXP;
#pragma unroll
    for (int e = 0; e < NEXP; e++) lg[e] += xv * rwr[e];
  }
  float mx = lg[0];
#pragma unroll
  for (int e = 1; e < NEXP; e++) mx = fmaxf(mx, lg[e]);
  float s = 0.f;
#pragma unroll
  for (int e = 0; e < NEXP; e++) { lg[e] = expf(lg[e] - mx); s += lg[e]; }
  const float inv = 1.f / s;
#pragma unroll
  for (int e = 0; e < NEXP; e++) lg[e] *= inv;
  int i1 = 0;
#pragma unroll
  for (int e = 1; e < NEXP; e++) if (lg[e] > lg[i1]) i1 = e;
  int i2 = (i1 == 0) ? 1 : 0;
#pragma unroll
  for (int e = 0; e < NEXP; e++) if (e != i1 && lg[e] > lg[i2]) i2 = e;
  float* gr = gates + t * NEXP;
#pragma unroll
  for (int e = 0; e < NEXP; e++)
    gr[e] = (e == i1) ? lg[i1] : ((e == i2) ? lg[i2] : 0.f);
  atomicAdd(&frac[i1], 1.f);
  atomicAdd(&frac[i2], 1.f);
#pragma unroll
  for (int e = 0; e < NEXP; e++) atomicAdd(&imp[e], lg[e]);
}

// ---------------- exact GELU fp32 -> bf16
__global__ void k_gelu(const float* __restrict__ h, __bf16* __restrict__ hb,
                       long long n) {
  long long i = (long long)blockIdx.x * blockDim.x + threadIdx.x;
  if (i >= n) return;
  float x = h[i];
  hb[i] = f2bf(0.5f * x * (1.f + erff(x * 0.70710678118654752f)));
}

// ---------------- fv = mean over tokens
__global__ void k_meantok(const float* __restrict__ o2, float* __restrict__ fv) {
  int i = blockIdx.x * blockDim.x + threadIdx.x;
  if (i >= NBATCH * D_MODEL) return;
  int b = i >> 9, d = i & 511;
  float s = 0.f;
  for (int ss = 0; ss < SEQ; ss++) s += o2[((long long)b * SEQ + ss) * D_MODEL + d];
  fv[i] = s * (1.f / 256.f);
}

// ---------------- classifier (tiny; scalar)
__global__ void k_cls(const float* __restrict__ fv, const float* __restrict__ w,
                      const float* __restrict__ bias, float* __restrict__ logits) {
  int i = blockIdx.x * blockDim.x + threadIdx.x;
  if (i >= NBATCH * D_MODEL) return;
  int b = i >> 9, n = i & 511;
  const float* f = fv + b * D_MODEL;
  float s = bias[n];
  for (int d = 0; d < D_MODEL; d++) s += f[d] * w[d * D_MODEL + n];
  logits[i] = s;
}

// ---------------- aux loss: acc = [frac1(8), imp1(8), frac2(8), imp2(8)]
__global__ void k_loss(const float* __restrict__ acc, float* __restrict__ out) {
  if (threadIdx.x == 0 && blockIdx.x == 0) {
    const float invT = 1.f / (float)T_TOK;
    float l1 = 0.f, l2 = 0.f;
    for (int e = 0; e < NEXP; e++) l1 += (acc[e] * invT) * (acc[8 + e] * invT);
    for (int e = 0; e < NEXP; e++) l2 += (acc[16 + e] * invT) * (acc[24 + e] * invT);
    out[0] = (float)NEXP * (l1 + l2);
  }
}

// ===========================================================================
extern "C" void kernel_launch(void* const* d_in, const int* in_sizes, int n_in,
                              void* d_out, int out_size, void* d_ws, size_t ws_size,
                              hipStream_t stream) {
  (void)in_sizes; (void)n_in; (void)out_size; (void)ws_size;
  const float* x    = (const float*)d_in[0];
  const float* pe_w = (const float*)d_in[1];
  const float* pe_b = (const float*)d_in[2];
  const float* pos  = (const float*)d_in[3];
  const float* ln1g = (const float*)d_in[4];  const float* ln1b = (const float*)d_in[5];
  const float* ln2g = (const float*)d_in[6];  const float* ln2b = (const float*)d_in[7];
  const float* ln3g = (const float*)d_in[8];  const float* ln3b = (const float*)d_in[9];
  const float* wq = (const float*)d_in[10]; const float* bq = (const float*)d_in[11];
  const float* wk = (const float*)d_in[12]; const float* bk = (const float*)d_in[13];
  const float* wv = (const float*)d_in[14]; const float* bv = (const float*)d_in[15];
  const float* wo = (const float*)d_in[16]; const float* bo = (const float*)d_in[17];
  const float* m1_rw = (const float*)d_in[18]; const float* m1_rb = (const float*)d_in[19];
  const float* m1_w1 = (const float*)d_in[20]; const float* m1_b1 = (const float*)d_in[21];
  const float* m1_w2 = (const float*)d_in[22]; const float* m1_b2 = (const float*)d_in[23];
  const float* m2_rw = (const float*)d_in[24]; const float* m2_rb = (const float*)d_in[25];
  const float* m2_w1 = (const float*)d_in[26]; const float* m2_b1 = (const float*)d_in[27];
  const float* m2_w2 = (const float*)d_in[28]; const float* m2_b2 = (const float*)d_in[29];
  const float* cls_w = (const float*)d_in[30]; const float* cls_b = (const float*)d_in[31];

  char* ws = (char*)d_ws;
  size_t cur = 0;
  auto alloc = [&](size_t bytes) -> char* {
    char* p = ws + cur;
    cur += (bytes + 255) & ~(size_t)255;
    return p;
  };

  __bf16* patches = (__bf16*)alloc((size_t)T_TOK * PDP * 2);
  __bf16* pe_wT   = (__bf16*)alloc((size_t)D_MODEL * PDP * 2);
  __bf16* wqT = (__bf16*)alloc((size_t)D_MODEL * D_MODEL * 2);
  __bf16* wkT = (__bf16*)alloc((size_t)D_MODEL * D_MODEL * 2);
  __bf16* wvT = (__bf16*)alloc((size_t)D_MODEL * D_MODEL * 2);
  __bf16* woT = (__bf16*)alloc((size_t)D_MODEL * D_MODEL * 2);
  __bf16* w1T1 = (__bf16*)alloc((size_t)NEXP * HID_DIM * D_MODEL * 2);
  __bf16* w2T1 = (__bf16*)alloc((size_t)NEXP * D_MODEL * HID_DIM * 2);
  __bf16* w1T2 = (__bf16*)alloc((size_t)NEXP * HID_DIM * D_MODEL * 2);
  __bf16* w2T2 = (__bf16*)alloc((size_t)NEXP * D_MODEL * HID_DIM * 2);
  float* e    = (float*)alloc((size_t)T_TOK * D_MODEL * 4);
  __bf16* lnb = (__bf16*)alloc((size_t)T_TOK * D_MODEL * 2);
  float* Qb = (float*)alloc((size_t)T_TOK * D_MODEL * 4);
  float* Kb = (float*)alloc((size_t)T_TOK * D_MODEL * 4);
  float* Vb = (float*)alloc((size_t)T_TOK * D_MODEL * 4);
  __bf16* qa = (__bf16*)alloc((size_t)BHEADS * SEQ * HD * 2);
  __bf16* ka = (__bf16*)alloc((size_t)BHEADS * SEQ * HD * 2);
  __bf16* vt = (__bf16*)alloc((size_t)BHEADS * HD * SEQ * 2);
  float* scores = (float*)alloc((size_t)BHEADS * SEQ * SEQ * 4);   // reused as h
  __bf16* attnb = (__bf16*)alloc((size_t)BHEADS * SEQ * SEQ * 2);  // reused as hb
  float* o1 = (float*)alloc((size_t)T_TOK * D_MODEL * 4);
  float* o2 = (float*)alloc((size_t)T_TOK * D_MODEL * 4);
  float* gates1 = (float*)alloc((size_t)T_TOK * NEXP * 4);
  float* gates2 = (float*)alloc((size_t)T_TOK * NEXP * 4);
  float* accbuf = (float*)alloc(32 * 4);
  // buffer reuse (lifetimes disjoint):
  float* ao = Qb;              // attention output, after Q is consumed
  __bf16* merged = (__bf16*)Kb;
  float* hbuf = scores;        // expert hidden fp32 (8192x2048) == scores size
  __bf16* hb = attnb;          // expert hidden bf16

  float* logits = (float*)d_out;
  float* fvout  = (float*)d_out + 16384;
  float* lossout = (float*)d_out + 32768;
  float* attnw  = (float*)d_out + 32769;

  const long long nTD = (long long)T_TOK * D_MODEL;

  // ---- weight conversions (transpose to N-major bf16, K padded) ----
  {
    long long n = (long long)D_MODEL * PDP;
    k_wtrans<<<(unsigned)((n + 255) / 256), 256, 0, stream>>>(pe_w, pe_wT, PD, D_MODEL, PDP, n);
    n = (long long)D_MODEL * D_MODEL;
    k_wtrans<<<(unsigned)((n + 255) / 256), 256, 0, stream>>>(wq, wqT, 512, 512, 512, n);
    k_wtrans<<<(unsigned)((n + 255) / 256), 256, 0, stream>>>(wk, wkT, 512, 512, 512, n);
    k_wtrans<<<(unsigned)((n + 255) / 256), 256, 0, stream>>>(wv, wvT, 512, 512, 512, n);
    k_wtrans<<<(unsigned)((n + 255) / 256), 256, 0, stream>>>(wo, woT, 512, 512, 512, n);
    n = (long long)NEXP * HID_DIM * D_MODEL;
    k_wtrans<<<(unsigned)((n + 255) / 256), 256, 0, stream>>>(m1_w1, w1T1, 512, 2048, 512, n);
    k_wtrans<<<(unsigned)((n + 255) / 256), 256, 0, stream>>>(m1_w2, w2T1, 2048, 512, 2048, n);
    k_wtrans<<<(unsigned)((n + 255) / 256), 256, 0, stream>>>(m2_w1, w1T2, 512, 2048, 512, n);
    k_wtrans<<<(unsigned)((n + 255) / 256), 256, 0, stream>>>(m2_w2, w2T2, 2048, 512, 2048, n);
  }

  // ---- patch embed: e = patches @ pe_w + pe_b + pos ----
  {
    long long n = (long long)T_TOK * PDP;
    k_patchify<<<(unsigned)((n + 255) / 256), 256, 0, stream>>>(x, patches);
  }
  k_gemm_bf16<<<dim3(32, 8, 1), 256, 0, stream>>>(patches, 0, PDP, pe_wT, 0, PDP,
      e, 0, 512, pe_b, 0, nullptr, 0, 1.f, 0, PDP);
  k_addpos<<<(unsigned)((nTD + 255) / 256), 256, 0, stream>>>(e, pos);

  // ---- attention ----
  k_ln<<<T_TOK / 8, 256, 0, stream>>>(e, ln1g, ln1b, lnb);
  k_gemm_bf16<<<dim3(32, 8, 1), 256, 0, stream>>>(lnb, 0, 512, wqT, 0, 512,
      Qb, 0, 512, bq, 0, nullptr, 0, 1.f, 0, 512);
  k_gemm_bf16<<<dim3(32, 8, 1), 256, 0, stream>>>(lnb, 0, 512, wkT, 0, 512,
      Kb, 0, 512, bk, 0, nullptr, 0, 1.f, 0, 512);
  k_gemm_bf16<<<dim3(32, 8, 1), 256, 0, stream>>>(lnb, 0, 512, wvT, 0, 512,
      Vb, 0, 512, bv, 0, nullptr, 0, 1.f, 0, 512);
  k_split<<<(unsigned)((nTD + 255) / 256), 256, 0, stream>>>(Qb, Kb, Vb, qa, ka, vt);
  // scores = (q @ k^T) / 8, batched over B*H
  k_gemm_bf16<<<dim3(1, 4, BHEADS), 256, 0, stream>>>(qa, (long long)SEQ * HD, HD,
      ka, (long long)SEQ * HD, HD, scores, (long long)SEQ * SEQ, SEQ,
      nullptr, 0, nullptr, 0, 0.125f, 0, HD);
  k_softmax<<<BHEADS * SEQ / 8, 256, 0, stream>>>(scores, attnb);
  k_attnw<<<NBATCH * SEQ / 8, 256, 0, stream>>>(scores, attnw);
  // attn @ v
  k_gemm_bf16<<<dim3(1, 1, BHEADS), 256, 0, stream>>>(attnb, (long long)SEQ * SEQ, SEQ,
      vt, (long long)HD * SEQ, SEQ, ao, (long long)SEQ * HD, HD,
      nullptr, 0, nullptr, 0, 1.f, 0, SEQ);
  k_merge<<<(unsigned)((nTD + 255) / 256), 256, 0, stream>>>(ao, merged);
  // out proj, accumulate residual into e
  k_gemm_bf16<<<dim3(32, 8, 1), 256, 0, stream>>>(merged, 0, 512, woT, 0, 512,
      e, 0, 512, bo, 0, nullptr, 0, 1.f, 1, 512);

  // ---- aux-loss accumulators ----
  k_zero<<<1, 32, 0, stream>>>(accbuf, 32);

  // ---- MoE layer 1 ----
  k_ln<<<T_TOK / 8, 256, 0, stream>>>(e, ln2g, ln2b, lnb);
  k_router<<<T_TOK / 256, 256, 0, stream>>>(lnb, m1_rw, m1_rb, gates1, accbuf, accbuf + 8);
  k_zero<<<(unsigned)((nTD + 255) / 256), 256, 0, stream>>>(o1, nTD);
  for (int ex = 0; ex < NEXP; ex++) {
    k_gemm_bf16<<<dim3(32, 32, 1), 256, 0, stream>>>(lnb, 0, 512,
        w1T1 + (size_t)ex * HID_DIM * D_MODEL, 0, 512, hbuf, 0, 2048,
        m1_b1 + ex * HID_DIM, 0, nullptr, 0, 1.f, 0, 512);
    long long nh = (long long)T_TOK * HID_DIM;
    k_gelu<<<(unsigned)((nh + 255) / 256), 256, 0, stream>>>(hbuf, hb, nh);
    k_gemm_bf16<<<dim3(32, 8, 1), 256, 0, stream>>>(hb, 0, 2048,
        w2T1 + (size_t)ex * D_MODEL * HID_DIM, 0, 2048, o1, 0, 512,
        m1_b2 + ex * D_MODEL, 0, gates1 + ex, NEXP, 1.f, 1, 2048);
  }

  // ---- MoE layer 2 ----
  k_ln<<<T_TOK / 8, 256, 0, stream>>>(o1, ln3g, ln3b, lnb);
  k_router<<<T_TOK / 256, 256, 0, stream>>>(lnb, m2_rw, m2_rb, gates2, accbuf + 16, accbuf + 24);
  k_zero<<<(unsigned)((nTD + 255) / 256), 256, 0, stream>>>(o2, nTD);
  for (int ex = 0; ex < NEXP; ex++) {
    k_gemm_bf16<<<dim3(32, 32, 1), 256, 0, stream>>>(lnb, 0, 512,
        w1T2 + (size_t)ex * HID_DIM * D_MODEL, 0, 512, hbuf, 0, 2048,
        m2_b1 + ex * HID_DIM, 0, nullptr, 0, 1.f, 0, 512);
    long long nh = (long long)T_TOK * HID_DIM;
    k_gelu<<<(unsigned)((nh + 255) / 256), 256, 0, stream>>>(hbuf, hb, nh);
    k_gemm_bf16<<<dim3(32, 8, 1), 256, 0, stream>>>(hb, 0, 2048,
        w2T2 + (size_t)ex * D_MODEL * HID_DIM, 0, 2048, o2, 0, 512,
        m2_b2 + ex * D_MODEL, 0, gates2 + ex, NEXP, 1.f, 1, 2048);
  }

  // ---- head ----
  k_meantok<<<(NBATCH * D_MODEL + 255) / 256, 256, 0, stream>>>(o2, fvout);
  k_cls<<<(NBATCH * D_MODEL + 255) / 256, 256, 0, stream>>>(fvout, cls_w, cls_b, logits);
  k_loss<<<1, 1, 0, stream>>>(accbuf, lossout);
}
